// DBHead_64072322122116
// MI455X (gfx1250) — matrix operand order repaired and checked by reading, not compiled, not run on hardware
//
#include <hip/hip_runtime.h>
#include <hip/hip_bf16.h>

typedef __bf16 bf16_t;
typedef __attribute__((ext_vector_type(16))) __bf16 v16bf;
typedef __attribute__((ext_vector_type(8)))  __bf16 v8bf;
typedef __attribute__((ext_vector_type(8)))  float  v8f;

#define NB   4
#define CIN  256
#define HH   64
#define WW   64
#define HWPX 4096      // 64*64
#define CQ   64
#define CR   64
#define H2PX 16384     // 128*128
#define OPIX 65536     // 256*256
#define BN_EPS 1e-5f

__device__ __forceinline__ v8f zero8() {
    v8f z;
#pragma unroll
    for (int i = 0; i < 8; ++i) z[i] = 0.0f;
    return z;
}

__device__ __forceinline__ v16bf load_a16(const bf16_t* lo, const bf16_t* hi) {
    const int hf = (threadIdx.x >> 4) & 1;
    v8bf l = *(const v8bf*)(lo + 8 * hf);
    v8bf h = *(const v8bf*)(lo + 8 * hf + 8);
    v16bf a;
#pragma unroll
    for (int i = 0; i < 8; ++i) { a[i] = l[i]; a[i + 8] = h[i]; }
    return a;
}

__device__ __forceinline__ v8f wmma_bf16(v16bf a, v16bf b, v8f c) {
    return __builtin_amdgcn_wmma_f32_16x16x32_bf16(false, a, false, b, (short)0, c, false, false);
}

// ---------------------------------------------------------------- repack kernels

// flat f32 -> bf16
__global__ void k_cvt_bf16(const float* __restrict__ in, bf16_t* __restrict__ out, int count) {
    int idx = blockIdx.x * blockDim.x + threadIdx.x;
    if (idx < count) out[idx] = (bf16_t)in[idx];
}

// x [N,C,HW] f32 -> xb [N,HW,C] bf16 (pixel-major for A fragments)
__global__ void k_xpose_bf16(const float* __restrict__ x, bf16_t* __restrict__ xb) {
    int idx = blockIdx.x * blockDim.x + threadIdx.x;
    if (idx >= NB * HWPX * CIN) return;
    int n = idx / (HWPX * CIN);
    int rem = idx - n * (HWPX * CIN);
    int p = rem / CIN;
    int c = rem - p * CIN;
    xb[idx] = (bf16_t)x[((size_t)n * CIN + c) * HWPX + p];
}

// cw [64,256,3,3] f32 -> w3 [64][tap*256+ci] bf16
__global__ void k_rep_w3(const float* __restrict__ cw, bf16_t* __restrict__ w3) {
    int idx = blockIdx.x * blockDim.x + threadIdx.x;
    if (idx >= CR * 2304) return;
    int co = idx / 2304;
    int k = idx - co * 2304;
    int tap = k >> 8;
    int ci = k & 255;
    w3[idx] = (bf16_t)cw[((size_t)co * CIN + ci) * 9 + tap];
}

// dw1 [2,2,64ci,64co] f32 -> wt [par][co][ci] bf16, kernel mirrored (jax conv_transpose)
__global__ void k_rep_wt1(const float* __restrict__ dw, bf16_t* __restrict__ wt) {
    int idx = blockIdx.x * blockDim.x + threadIdx.x;
    if (idx >= 4 * CR * CR) return;
    int par = idx >> 12;
    int rem = idx & 4095;
    int co = rem >> 6;
    int ci = rem & 63;
    wt[idx] = (bf16_t)dw[(size_t)(3 - par) * 4096 + ci * 64 + co];
}

// ---------------------------------------------------------------- 1x1 conv GEMM + bias + PReLU
// A: xb [N,HW,256] bf16.  Wt: [Cout,256] bf16.  out mode 0: [N,HW,Cout]  mode 1: [N,Cout,HW]
__global__ void k_gemm1x1(const bf16_t* __restrict__ A, const bf16_t* __restrict__ Wt,
                          const float* __restrict__ bias, const float* __restrict__ alpha,
                          bf16_t* __restrict__ out, int Cout, int mode) {
    const int lane = threadIdx.x;
    const int half = lane >> 4;
    const int lm   = lane & 15;
    const int p0   = blockIdx.x * 16;
    const int cob  = blockIdx.y * 64;
    const int n    = blockIdx.z;

    const bf16_t* Arow = A + ((size_t)n * HWPX + p0 + lm) * CIN;
    v8f acc[4];
#pragma unroll
    for (int t = 0; t < 4; ++t) acc[t] = zero8();

#pragma unroll
    for (int c0 = 0; c0 < CIN; c0 += 32) {
        v16bf a = load_a16(Arow + c0 + 8 * half, Arow + c0 + 16 + 8 * half);
#pragma unroll
        for (int t = 0; t < 4; ++t) {
            int co = cob + t * 16 + lm;
            v16bf b = *(const v16bf*)(Wt + (size_t)co * CIN + c0 + 16 * half);
            acc[t] = wmma_bf16(a, b, acc[t]);
        }
    }
    const float al = alpha[0];
#pragma unroll
    for (int t = 0; t < 4; ++t) {
        int co = cob + t * 16 + lm;
        float bs = bias[co];
#pragma unroll
        for (int r = 0; r < 8; ++r) {
            int p = p0 + r + 8 * half;
            float v = acc[t][r] + bs;
            v = (v >= 0.0f) ? v : al * v;
            if (mode == 0) out[((size_t)n * HWPX + p) * Cout + co] = (bf16_t)v;
            else           out[((size_t)n * Cout + co) * HWPX + p] = (bf16_t)v;
        }
    }
}

// ---------------------------------------------------------------- attention pass 1: row max & sumexp
// Q,K: [N,HW,64] bf16 pixel-major.  mbuf,lbuf: [N*HW] f32
__global__ void k_attn_stats(const bf16_t* __restrict__ Q, const bf16_t* __restrict__ K,
                             float* __restrict__ mbuf, float* __restrict__ lbuf) {
    __shared__ float red[32][16];
    const int lane = threadIdx.x;
    const int half = lane >> 4;
    const int lm   = lane & 15;
    const int p0   = blockIdx.x * 16;
    const int n    = blockIdx.y;

    const bf16_t* qrow = Q + ((size_t)n * HWPX + p0 + lm) * CQ;
    v16bf qa0 = load_a16(qrow + 8 * half,      qrow + 16 + 8 * half);
    v16bf qa1 = load_a16(qrow + 32 + 8 * half, qrow + 48 + 8 * half);

    float mx[8], sx[8];
#pragma unroll
    for (int r = 0; r < 8; ++r) { mx[r] = -3.0e38f; sx[r] = 0.0f; }

    const bf16_t* Kn = K + (size_t)n * HWPX * CQ;
    for (int key0 = 0; key0 < HWPX; key0 += 16) {
        const bf16_t* krow = Kn + (size_t)(key0 + lm) * CQ;
        v16bf b0 = *(const v16bf*)(krow + 16 * half);
        v16bf b1 = *(const v16bf*)(krow + 32 + 16 * half);
        v8f s = zero8();
        s = wmma_bf16(qa0, b0, s);
        s = wmma_bf16(qa1, b1, s);
#pragma unroll
        for (int r = 0; r < 8; ++r) {
            float v = s[r];
            if (v > mx[r]) { sx[r] = sx[r] * __expf(mx[r] - v) + 1.0f; mx[r] = v; }
            else           { sx[r] += __expf(v - mx[r]); }
        }
    }
#pragma unroll
    for (int r = 0; r < 8; ++r) { red[lane][r] = mx[r]; red[lane][8 + r] = sx[r]; }
    __syncthreads();
    if (lane < 16) {
        int r  = lane & 7;
        int L0 = (lane < 8) ? 0 : 16;
        float M = -3.0e38f;
        for (int i = 0; i < 16; ++i) M = fmaxf(M, red[L0 + i][r]);
        float L = 0.0f;
        for (int i = 0; i < 16; ++i) L += red[L0 + i][8 + r] * __expf(red[L0 + i][r] - M);
        mbuf[(size_t)n * HWPX + p0 + lane] = M;
        lbuf[(size_t)n * HWPX + p0 + lane] = L;
    }
}

// ---------------------------------------------------------------- attention pass 2: softmax(QK) @ V
// V: asm [N,256,HW] bf16 channel-major.  xnb out: [N,256,HW] bf16
__global__ void k_attn_out(const bf16_t* __restrict__ Q, const bf16_t* __restrict__ K,
                           const bf16_t* __restrict__ V,
                           const float* __restrict__ mbuf, const float* __restrict__ lbuf,
                           bf16_t* __restrict__ xnb) {
    __shared__ bf16_t pt[16 * 32];
    const int lane = threadIdx.x;
    const int half = lane >> 4;
    const int lm   = lane & 15;
    const int p0   = blockIdx.x * 16;
    const int cob  = blockIdx.y * 64;
    const int n    = blockIdx.z;

    const bf16_t* qrow = Q + ((size_t)n * HWPX + p0 + lm) * CQ;
    v16bf qa0 = load_a16(qrow + 8 * half,      qrow + 16 + 8 * half);
    v16bf qa1 = load_a16(qrow + 32 + 8 * half, qrow + 48 + 8 * half);

    float rm[8], rl[8];
#pragma unroll
    for (int r = 0; r < 8; ++r) {
        int row = p0 + r + 8 * half;
        rm[r] = mbuf[(size_t)n * HWPX + row];
        rl[r] = 1.0f / lbuf[(size_t)n * HWPX + row];
    }

    v8f acc[4];
#pragma unroll
    for (int t = 0; t < 4; ++t) acc[t] = zero8();

    const bf16_t* Kn = K + (size_t)n * HWPX * CQ;
    const bf16_t* Vn = V + (size_t)n * CIN * HWPX;

    for (int key0 = 0; key0 < HWPX; key0 += 32) {
#pragma unroll
        for (int t = 0; t < 2; ++t) {
            const bf16_t* krow = Kn + (size_t)(key0 + t * 16 + lm) * CQ;
            v16bf b0 = *(const v16bf*)(krow + 16 * half);
            v16bf b1 = *(const v16bf*)(krow + 32 + 16 * half);
            v8f s = zero8();
            s = wmma_bf16(qa0, b0, s);
            s = wmma_bf16(qa1, b1, s);
#pragma unroll
            for (int r = 0; r < 8; ++r)
                pt[(r + 8 * half) * 32 + t * 16 + lm] = (bf16_t)__expf(s[r] - rm[r]);
        }
        __syncthreads();
        v16bf pa = load_a16(&pt[lm * 32 + 8 * half], &pt[lm * 32 + 16 + 8 * half]);
#pragma unroll
        for (int t = 0; t < 4; ++t) {
            int co = cob + t * 16 + lm;
            v16bf bv = *(const v16bf*)(Vn + (size_t)co * HWPX + key0 + 16 * half);
            acc[t] = wmma_bf16(pa, bv, acc[t]);
        }
        __syncthreads();
    }
#pragma unroll
    for (int t = 0; t < 4; ++t) {
        int co = cob + t * 16 + lm;
#pragma unroll
        for (int r = 0; r < 8; ++r) {
            int p = p0 + r + 8 * half;
            xnb[((size_t)n * CIN + co) * HWPX + p] = (bf16_t)(acc[t][r] * rl[r]);
        }
    }
}

// ---------------------------------------------------------------- conv3x3 (256->64, SAME) + BN + ReLU
// in: [N,256,HW] bf16 channel-major.  w3: [64][tap*256+ci] bf16.  out: [N,HW,64] bf16 pixel-major
__global__ void k_conv3x3(const bf16_t* __restrict__ in, const bf16_t* __restrict__ w3,
                          const float* __restrict__ g, const float* __restrict__ b,
                          const float* __restrict__ mn, const float* __restrict__ vr,
                          bf16_t* __restrict__ out) {
    __shared__ bf16_t patch[3 * 18 * 32];
    const int lane = threadIdx.x;
    const int half = lane >> 4;
    const int lm   = lane & 15;
    const int p0   = blockIdx.x * 16;
    const int n    = blockIdx.z;
    const int h    = p0 >> 6;
    const int w0   = p0 & 63;

    v8f acc[4];
#pragma unroll
    for (int t = 0; t < 4; ++t) acc[t] = zero8();

    const bf16_t* inb = in + (size_t)n * CIN * HWPX;
    for (int c8 = 0; c8 < 8; ++c8) {
        int ci0 = c8 * 32;
        for (int idx = lane; idx < 3 * 18 * 32; idx += 32) {
            int ci   = idx & 31;
            int rest = idx >> 5;
            int wx   = rest % 18;
            int dy   = rest / 18;
            int hr   = h + dy - 1;
            int wc   = w0 + wx - 1;
            bf16_t val = (bf16_t)0.0f;
            if (hr >= 0 && hr < HH && wc >= 0 && wc < WW)
                val = inb[(size_t)(ci0 + ci) * HWPX + hr * WW + wc];
            patch[idx] = val;
        }
        __syncthreads();
#pragma unroll
        for (int tap = 0; tap < 9; ++tap) {
            int dy = tap / 3, dx = tap % 3;
            const bf16_t* prow = &patch[(dy * 18 + (lm + dx)) * 32];
            v16bf a = load_a16(prow + 8 * half, prow + 16 + 8 * half);
#pragma unroll
            for (int t = 0; t < 4; ++t) {
                int co = t * 16 + lm;
                v16bf bb = *(const v16bf*)(w3 + (size_t)co * 2304 + tap * 256 + ci0 + 16 * half);
                acc[t] = wmma_bf16(a, bb, acc[t]);
            }
        }
        __syncthreads();
    }
#pragma unroll
    for (int t = 0; t < 4; ++t) {
        int co = t * 16 + lm;
        float inv  = g[co] * rsqrtf(vr[co] + BN_EPS);
        float beta = b[co] - mn[co] * inv;
#pragma unroll
        for (int r = 0; r < 8; ++r) {
            int p = p0 + r + 8 * half;
            float x = acc[t][r] * inv + beta;
            x = fmaxf(x, 0.0f);
            out[((size_t)n * HWPX + p) * CR + co] = (bf16_t)x;
        }
    }
}

// ---------------------------------------------------------------- convT 2x2 s2 (64->64) + bias + BN + ReLU
// in: [N,4096,64] bf16.  wt: [par][co][ci] bf16.  out: [N,16384,64] bf16 pixel-major
__global__ void k_convT1(const bf16_t* __restrict__ in, const bf16_t* __restrict__ wt,
                         const float* __restrict__ db,
                         const float* __restrict__ g, const float* __restrict__ b,
                         const float* __restrict__ mn, const float* __restrict__ vr,
                         bf16_t* __restrict__ out) {
    const int lane = threadIdx.x;
    const int half = lane >> 4;
    const int lm   = lane & 15;
    const int p0   = blockIdx.x * 16;
    const int n    = blockIdx.z;

    const bf16_t* arow = in + ((size_t)n * HWPX + p0 + lm) * CR;
    v16bf a0 = load_a16(arow + 8 * half,      arow + 16 + 8 * half);
    v16bf a1 = load_a16(arow + 32 + 8 * half, arow + 48 + 8 * half);

    for (int par = 0; par < 4; ++par) {
        int dy = par >> 1, dx = par & 1;
        v8f acc[4];
#pragma unroll
        for (int t = 0; t < 4; ++t) acc[t] = zero8();
#pragma unroll
        for (int t = 0; t < 4; ++t) {
            int co = t * 16 + lm;
            const bf16_t* wrow = wt + ((size_t)par * 64 + co) * 64;
            v16bf b0 = *(const v16bf*)(wrow + 16 * half);
            v16bf b1 = *(const v16bf*)(wrow + 32 + 16 * half);
            acc[t] = wmma_bf16(a0, b0, acc[t]);
            acc[t] = wmma_bf16(a1, b1, acc[t]);
        }
#pragma unroll
        for (int t = 0; t < 4; ++t) {
            int co = t * 16 + lm;
            float inv  = g[co] * rsqrtf(vr[co] + BN_EPS);
            float beta = b[co] - mn[co] * inv;
            float bs   = db[co];
#pragma unroll
            for (int r = 0; r < 8; ++r) {
                int p  = p0 + r + 8 * half;
                int hh = p >> 6, ww = p & 63;
                int opix = (2 * hh + dy) * 128 + 2 * ww + dx;
                float x = (acc[t][r] + bs) * inv + beta;
                x = fmaxf(x, 0.0f);
                out[((size_t)n * H2PX + opix) * CR + co] = (bf16_t)x;
            }
        }
    }
}

// ---------------------------------------------------------------- convT 2x2 s2 (64->1) + bias + sigmoid
// in: [N,16384,64] bf16.  w2: dw2 [2,2,64,1] f32.  map: [N,65536] f32
__global__ void k_convT2_sig(const bf16_t* __restrict__ in, const float* __restrict__ w2,
                             const float* __restrict__ b2, float* __restrict__ map) {
    int idx = blockIdx.x * blockDim.x + threadIdx.x;
    if (idx >= NB * OPIX) return;
    int n   = idx >> 16;
    int pix = idx & 65535;
    int h2 = pix >> 8, w2i = pix & 255;
    int hh = h2 >> 1, ww = w2i >> 1, dy = h2 & 1, dx = w2i & 1;
    const bf16_t* row = in + ((size_t)n * H2PX + hh * 128 + ww) * CR;
    const float* wtap = w2 + (3 - (dy * 2 + dx)) * 64;  // mirrored (jax conv_transpose)
    float acc = b2[0];
#pragma unroll 8
    for (int ci = 0; ci < 64; ++ci) acc += (float)row[ci] * wtap[ci];
    map[idx] = 1.0f / (1.0f + __expf(-acc));
}

// ---------------------------------------------------------------- diff-binarize + concat
__global__ void k_combine(const float* __restrict__ prob, const float* __restrict__ thr,
                          float* __restrict__ out) {
    int idx = blockIdx.x * blockDim.x + threadIdx.x;
    if (idx >= NB * OPIX) return;
    int n = idx >> 16;
    int p = idx & 65535;
    float pv = prob[idx], tv = thr[idx];
    size_t base = (size_t)n * 3 * OPIX;
    out[base + p]             = pv;
    out[base + OPIX + p]      = tv;
    out[base + 2 * OPIX + p]  = 1.0f / (1.0f + __expf(-50.0f * (pv - tv)));
}

// ---------------------------------------------------------------- launch

extern "C" void kernel_launch(void* const* d_in, const int* in_sizes, int n_in,
                              void* d_out, int out_size, void* d_ws, size_t ws_size,
                              hipStream_t stream) {
    (void)in_sizes; (void)n_in; (void)out_size; (void)ws_size;

    const float* x     = (const float*)d_in[0];
    const float* wm1   = (const float*)d_in[1];
    const float* bm1   = (const float*)d_in[2];
    const float* a1    = (const float*)d_in[3];
    const float* wm2   = (const float*)d_in[4];
    const float* bm2   = (const float*)d_in[5];
    const float* a2    = (const float*)d_in[6];
    const float* wa    = (const float*)d_in[7];
    const float* ba    = (const float*)d_in[8];
    const float* aa    = (const float*)d_in[9];
    const float* bz_cw = (const float*)d_in[10];
    const float* bz_g1 = (const float*)d_in[11];
    const float* bz_b1 = (const float*)d_in[12];
    const float* bz_m1 = (const float*)d_in[13];
    const float* bz_v1 = (const float*)d_in[14];
    const float* bz_dw1= (const float*)d_in[15];
    const float* bz_db1= (const float*)d_in[16];
    const float* bz_g2 = (const float*)d_in[17];
    const float* bz_b2 = (const float*)d_in[18];
    const float* bz_m2 = (const float*)d_in[19];
    const float* bz_v2 = (const float*)d_in[20];
    const float* bz_dw2= (const float*)d_in[21];
    const float* bz_db2= (const float*)d_in[22];
    const float* th_cw = (const float*)d_in[23];
    const float* th_g1 = (const float*)d_in[24];
    const float* th_b1 = (const float*)d_in[25];
    const float* th_m1 = (const float*)d_in[26];
    const float* th_v1 = (const float*)d_in[27];
    const float* th_dw1= (const float*)d_in[28];
    const float* th_db1= (const float*)d_in[29];
    const float* th_g2 = (const float*)d_in[30];
    const float* th_b2 = (const float*)d_in[31];
    const float* th_m2 = (const float*)d_in[32];
    const float* th_v2 = (const float*)d_in[33];
    const float* th_dw2= (const float*)d_in[34];
    const float* th_db2= (const float*)d_in[35];
    float* out = (float*)d_out;

    char* ws = (char*)d_ws;
    size_t off = 0;
    auto alloc = [&](size_t bytes) -> void* {
        void* p = ws + off;
        off = (off + bytes + 255) & ~(size_t)255;
        return p;
    };
    bf16_t* xb    = (bf16_t*)alloc((size_t)NB * HWPX * CIN * 2);   // pixel-major x
    bf16_t* xc    = (bf16_t*)alloc((size_t)NB * CIN * HWPX * 2);   // channel-major x
    bf16_t* w1b   = (bf16_t*)alloc((size_t)CQ * CIN * 2);
    bf16_t* w2b   = (bf16_t*)alloc((size_t)CQ * CIN * 2);
    bf16_t* wab   = (bf16_t*)alloc((size_t)CIN * CIN * 2);
    bf16_t* e1b   = (bf16_t*)alloc((size_t)NB * HWPX * CQ * 2);
    bf16_t* e2b   = (bf16_t*)alloc((size_t)NB * HWPX * CQ * 2);
    bf16_t* asmb  = (bf16_t*)alloc((size_t)NB * CIN * HWPX * 2);
    float*  mbuf  = (float*)alloc((size_t)NB * HWPX * 4);
    float*  lbuf  = (float*)alloc((size_t)NB * HWPX * 4);
    bf16_t* xnb   = (bf16_t*)alloc((size_t)NB * CIN * HWPX * 2);
    bf16_t* w3bz  = (bf16_t*)alloc((size_t)CR * 2304 * 2);
    bf16_t* w3th  = (bf16_t*)alloc((size_t)CR * 2304 * 2);
    bf16_t* wt1bz = (bf16_t*)alloc((size_t)4 * CR * CR * 2);
    bf16_t* wt1th = (bf16_t*)alloc((size_t)4 * CR * CR * 2);
    bf16_t* h1b   = (bf16_t*)alloc((size_t)NB * HWPX * CR * 2);    // shared by both branches
    bf16_t* up1b  = (bf16_t*)alloc((size_t)NB * H2PX * CR * 2);    // shared by both branches
    float*  prob  = (float*)alloc((size_t)NB * OPIX * 4);
    float*  thr   = (float*)alloc((size_t)NB * OPIX * 4);

    const int TPB = 256;
    // -------- prep / repack
    k_xpose_bf16<<<(NB * HWPX * CIN + TPB - 1) / TPB, TPB, 0, stream>>>(x, xb);
    k_cvt_bf16<<<(NB * CIN * HWPX + TPB - 1) / TPB, TPB, 0, stream>>>(x, xc, NB * CIN * HWPX);
    k_cvt_bf16<<<(CQ * CIN + TPB - 1) / TPB, TPB, 0, stream>>>(wm1, w1b, CQ * CIN);
    k_cvt_bf16<<<(CQ * CIN + TPB - 1) / TPB, TPB, 0, stream>>>(wm2, w2b, CQ * CIN);
    k_cvt_bf16<<<(CIN * CIN + TPB - 1) / TPB, TPB, 0, stream>>>(wa, wab, CIN * CIN);
    k_rep_w3<<<(CR * 2304 + TPB - 1) / TPB, TPB, 0, stream>>>(bz_cw, w3bz);
    k_rep_w3<<<(CR * 2304 + TPB - 1) / TPB, TPB, 0, stream>>>(th_cw, w3th);
    k_rep_wt1<<<(4 * CR * CR + TPB - 1) / TPB, TPB, 0, stream>>>(bz_dw1, wt1bz);
    k_rep_wt1<<<(4 * CR * CR + TPB - 1) / TPB, TPB, 0, stream>>>(th_dw1, wt1th);

    // -------- non-local attention
    k_gemm1x1<<<dim3(HWPX / 16, 1, NB), 32, 0, stream>>>(xb, w1b, bm1, a1, e1b, CQ, 0);
    k_gemm1x1<<<dim3(HWPX / 16, 1, NB), 32, 0, stream>>>(xb, w2b, bm2, a2, e2b, CQ, 0);
    k_gemm1x1<<<dim3(HWPX / 16, 4, NB), 32, 0, stream>>>(xb, wab, ba, aa, asmb, CIN, 1);
    k_attn_stats<<<dim3(HWPX / 16, NB), 32, 0, stream>>>(e1b, e2b, mbuf, lbuf);
    k_attn_out<<<dim3(HWPX / 16, 4, NB), 32, 0, stream>>>(e1b, e2b, asmb, mbuf, lbuf, xnb);

    // -------- binarize branch (on attention output)
    k_conv3x3<<<dim3(HWPX / 16, 1, NB), 32, 0, stream>>>(xnb, w3bz, bz_g1, bz_b1, bz_m1, bz_v1, h1b);
    k_convT1<<<dim3(HWPX / 16, 1, NB), 32, 0, stream>>>(h1b, wt1bz, bz_db1, bz_g2, bz_b2, bz_m2, bz_v2, up1b);
    k_convT2_sig<<<(NB * OPIX + TPB - 1) / TPB, TPB, 0, stream>>>(up1b, bz_dw2, bz_db2, prob);

    // -------- threshold branch (on raw x); reuses h1b/up1b (stream-ordered)
    k_conv3x3<<<dim3(HWPX / 16, 1, NB), 32, 0, stream>>>(xc, w3th, th_g1, th_b1, th_m1, th_v1, h1b);
    k_convT1<<<dim3(HWPX / 16, 1, NB), 32, 0, stream>>>(h1b, wt1th, th_db1, th_g2, th_b2, th_m2, th_v2, up1b);
    k_convT2_sig<<<(NB * OPIX + TPB - 1) / TPB, TPB, 0, stream>>>(up1b, th_dw2, th_db2, thr);

    // -------- diff binarize + concat
    k_combine<<<(NB * OPIX + TPB - 1) / TPB, TPB, 0, stream>>>(prob, thr, out);
}
